// CombinedGNNClassifier_31387620999576
// MI455X (gfx1250) — compile-verified, matrix operand order checked by reading, not run on hardware
//
#include <hip/hip_runtime.h>
#include <hip/hip_bf16.h>
#include <math.h>

// ============================================================================
// CombinedGNNClassifier for MI455X (gfx1250, wave32, WMMA).
//
// Roofline: ~90 GFLOP total; after this round the whole ~146MB workspace fits
// in the 192MB L2, so the pipeline is matrix-core bound. All GEMM-shaped math
// goes through v_wmma_f32_16x16x32_f16 with an f16-resident data path:
//   - weights pre-packed to fragment-major f16 (one v16h = 2x b128 per lane)
//   - activations produced/consumed as f16 row-major (2x b128 per A frag)
//   -> GEMM inner loop = 8 b128 loads + 4 WMMAs, zero cvt instructions.
// Attention is fused (QK^T -> LDS softmax -> PV) in f16.
//
// Input flattening assumption (jax pytree: top-level dict insertion order,
// nested dicts sorted by key, lists in order):
//   0 tm, 1 spec, 2 scat, 3 snr,
//   4..11   ae: d1b,d1w,d2b,d2w,e1b,e1w,e2b,e2w
//   12..37  fsk encoder (26 leaves, sorted; see load_enc)
//   38 g1b, 39 g1w, 40 g2b, 41 g2w
//   42..59  gnn[i]: att,bias,bl,br,wl,wr  (i=0..2)
//   60 h1b, 61 h1w, 62 h2b, 63 h2w
//   64..72  norms[i]: b,g,ms
//   73 pool_b, 74 pool_w, 75 pos
//   76..101 psk encoder, 102..127 qam encoder
//   128 snr_b, 129 snr_w
// ============================================================================

typedef __attribute__((ext_vector_type(16))) _Float16 v16h;
typedef __attribute__((ext_vector_type(8)))  _Float16 v8h;
typedef __attribute__((ext_vector_type(8)))  float    v8f;

#define ACT_NONE 0
#define ACT_RELU 1
#define ACT_GELU 2

#define WMMA_F16(a, b, c) \
  __builtin_amdgcn_wmma_f32_16x16x32_f16(false, (a), false, (b), (short)0, (c), false, false)

// ---------------------------------------------------------------------------
// WMMA fragment helpers (layouts per CDNA5 ISA 7.12.2: 16-bit 16x32 A frag,
// 32x16 B frag, 16x16 f32 C/D frag).
// Lane roles: l16 = lane&15 (A row / B col), hs = 8*(lane>=16).
// v16h element 2r+e holds K index kslot(r,hs)+e.
// ---------------------------------------------------------------------------
__device__ __forceinline__ int kslot(int r, int hs) {
  return (r < 4) ? (2 * r + hs) : (16 + 2 * (r - 4) + hs);
}

// A fragment from f32 row-major; p = per-lane row base.
__device__ __forceinline__ v16h fragA_f(const float* p, int k0, int hs) {
  v16h a;
#pragma unroll
  for (int r = 0; r < 8; ++r) {
    int k = k0 + kslot(r, hs);
    a[2 * r]     = (_Float16)p[k];
    a[2 * r + 1] = (_Float16)p[k + 1];
  }
  return a;
}

// A fragment from f16 row-major; two contiguous 16B runs -> 2x b128 loads.
__device__ __forceinline__ v16h fragA_h(const _Float16* p, int k0, int hs) {
  v8h lo = *(const v8h*)(p + k0 + hs);
  v8h hi = *(const v8h*)(p + k0 + 16 + hs);
  v16h a;
#pragma unroll
  for (int i = 0; i < 8; ++i) {
    a[i] = lo[i];
    a[8 + i] = hi[i];
  }
  return a;
}

// A fragment from LDS f16; prow = per-lane row base.
__device__ __forceinline__ v16h fragA_lds(const _Float16* prow, int k0, int hs) {
  v16h a;
#pragma unroll
  for (int r = 0; r < 8; ++r) {
    int k = k0 + kslot(r, hs);
    a[2 * r]     = prow[k];
    a[2 * r + 1] = prow[k + 1];
  }
  return a;
}

// B fragment from f16 row-major K x N (strided); p pre-offset to (0, col+l16).
__device__ __forceinline__ v16h fragB_rm_h(const _Float16* p, size_t ld, int hs) {
  v16h b;
#pragma unroll
  for (int r = 0; r < 8; ++r) {
    int k = kslot(r, hs);
    b[2 * r]     = p[(size_t)k * ld];
    b[2 * r + 1] = p[(size_t)(k + 1) * ld];
  }
  return b;
}

// Clamped f32 A fragment (edge GEMM): branch-free select-zero.
__device__ __forceinline__ v16h fragA_c(const float* A, int ld, int M, int K,
                                        int row0, int k0, int lane) {
  v16h a;
  int m = row0 + (lane & 15);
  float ms = (m < M) ? 1.f : 0.f;
  int mc = (m < M) ? m : (M - 1);
  const float* p = A + (size_t)mc * ld;
  int hs = (lane >> 4) << 3;
#pragma unroll
  for (int r = 0; r < 8; ++r) {
    int k = k0 + kslot(r, hs);
    int kc0 = (k < K) ? k : 0;
    int kc1 = (k + 1 < K) ? (k + 1) : 0;
    float f0 = p[kc0] * ((k < K) ? ms : 0.f);
    float f1 = p[kc1] * ((k + 1 < K) ? ms : 0.f);
    a[2 * r]     = (_Float16)f0;
    a[2 * r + 1] = (_Float16)f1;
  }
  return a;
}

// Clamped f32 B fragment (edge GEMM).
__device__ __forceinline__ v16h fragB_c(const float* B, int ld, int K, int N,
                                        int k0, int col0, int lane) {
  v16h b;
  int n = col0 + (lane & 15);
  float ns = (n < N) ? 1.f : 0.f;
  int nc = (n < N) ? n : (N - 1);
  int hs = (lane >> 4) << 3;
#pragma unroll
  for (int r = 0; r < 8; ++r) {
    int k = k0 + kslot(r, hs);
    int kc0 = (k < K) ? k : 0;
    int kc1 = (k + 1 < K) ? (k + 1) : 0;
    float f0 = B[(size_t)kc0 * ld + nc] * ((k < K) ? ns : 0.f);
    float f1 = B[(size_t)kc1 * ld + nc] * ((k + 1 < K) ? ns : 0.f);
    b[2 * r]     = (_Float16)f0;
    b[2 * r + 1] = (_Float16)f1;
  }
  return b;
}

__device__ __forceinline__ float apply_act(float v, int act) {
  if (act == ACT_RELU) return fmaxf(v, 0.f);
  if (act == ACT_GELU) return 0.5f * v * (1.f + erff(v * 0.70710678118654752f));
  return v;
}

__device__ __forceinline__ v16h loadA(const float* p, int k0, int hs) {
  return fragA_f(p, k0, hs);
}
__device__ __forceinline__ v16h loadA(const _Float16* p, int k0, int hs) {
  return fragA_h(p, k0, hs);
}

// ---------------------------------------------------------------------------
// Weight pre-pack: f32 (K x N) row-major -> fragment-major f16. The fragment
// for (ktile kt, ntile nt) is 32 lanes x 16 halves, laid out contiguously, so
// a GEMM B-frag load is a single aligned v16h (= 2x global_load_b128).
// Requires K%32==0, N%16==0.
// ---------------------------------------------------------------------------
__global__ void pack_b_kernel(const float* __restrict__ B, _Float16* __restrict__ out,
                              int K, int N) {
  int idx = blockIdx.x * blockDim.x + threadIdx.x;
  if (idx >= K * N) return;
  int n = idx % N, k = idx / N;
  int kt = k >> 5, kw = k & 31;
  int hi = (kw >> 3) & 1;
  int kk = kw - 8 * hi;  // 0..7 or 16..23
  int r = (kk < 8) ? (kk >> 1) : (4 + ((kk - 16) >> 1));
  int e = kk & 1;
  int nt = n >> 4;
  int lane = (n & 15) + 16 * hi;
  size_t off = (((size_t)kt * (N >> 4) + nt) * 32 + lane) * 16 + 2 * r + e;
  out[off] = (_Float16)B[idx];
}

// ---------------------------------------------------------------------------
// Fast GEMM: C = act(A(MxK) @ B(KxN) + bias) + residual, M%32==N%32==K%32==0.
// One wave per 32x32 tile. AT in {float,_Float16}, OT in {float,_Float16}.
// Bp is fragment-major packed f16. residual (optional) is f32 row-major.
// ---------------------------------------------------------------------------
template <typename AT, typename OT>
__launch_bounds__(32)
__global__ void gemm32p_kernel(const AT* __restrict__ A, const _Float16* __restrict__ Bp,
                               const float* __restrict__ bias, const float* residual,
                               OT* C, int M, int N, int K, int act) {
  const int lane = threadIdx.x;
  const int col0 = blockIdx.x * 32;
  const int row0 = blockIdx.y * 32;
  const int hs = (lane >> 4) << 3;
  const AT* a0p = A + (size_t)(row0 + (lane & 15)) * K;
  const AT* a1p = a0p + (size_t)16 * K;
  const _Float16* bp = Bp + (((size_t)(col0 >> 4)) * 32 + lane) * 16;
  const size_t kstride = (size_t)(N >> 4) * 512;  // halves per k-tile
  v8f acc00 = {}, acc01 = {}, acc10 = {}, acc11 = {};
  for (int k0 = 0; k0 < K; k0 += 32) {
    v16h a0 = loadA(a0p, k0, hs);
    v16h a1 = loadA(a1p, k0, hs);
    v16h b0 = *(const v16h*)bp;
    v16h b1 = *(const v16h*)(bp + 512);
    bp += kstride;
    acc00 = WMMA_F16(a0, b0, acc00);
    acc01 = WMMA_F16(a0, b1, acc01);
    acc10 = WMMA_F16(a1, b0, acc10);
    acc11 = WMMA_F16(a1, b1, acc11);
  }
  const int n = col0 + (lane & 15);
  const int hi = lane >> 4;
  const float bv0 = bias ? bias[n] : 0.f;
  const float bv1 = bias ? bias[n + 16] : 0.f;
#pragma unroll
  for (int r = 0; r < 8; ++r) {
    int m0 = row0 + r + 8 * hi;
    int m1 = m0 + 16;
    size_t i00 = (size_t)m0 * N + n;
    size_t i01 = i00 + 16;
    size_t i10 = (size_t)m1 * N + n;
    size_t i11 = i10 + 16;
    float v00 = apply_act(acc00[r] + bv0, act);
    float v01 = apply_act(acc01[r] + bv1, act);
    float v10 = apply_act(acc10[r] + bv0, act);
    float v11 = apply_act(acc11[r] + bv1, act);
    if (residual) {
      v00 += residual[i00];
      v01 += residual[i01];
      v10 += residual[i10];
      v11 += residual[i11];
    }
    C[i00] = (OT)v00;
    C[i01] = (OT)v01;
    C[i10] = (OT)v10;
    C[i11] = (OT)v11;
  }
}

// ---------------------------------------------------------------------------
// Edge GEMM for odd shapes (scat K=44, head M=8/N=24): clamped branch-free
// f32 fragment loads, one wave per 16x16 tile.
// ---------------------------------------------------------------------------
__launch_bounds__(32)
__global__ void gemm_edge_kernel(const float* __restrict__ A, const float* __restrict__ B,
                                 const float* __restrict__ bias, const float* residual,
                                 float* C, int M, int N, int K, int act) {
  const int lane = threadIdx.x;
  const int col0 = blockIdx.x * 16;
  const int row0 = blockIdx.y * 16;
  v8f acc = {};
  for (int k0 = 0; k0 < K; k0 += 32) {
    v16h a = fragA_c(A, K, M, K, row0, k0, lane);
    v16h b = fragB_c(B, N, K, N, k0, col0, lane);
    acc = WMMA_F16(a, b, acc);
  }
  int n = col0 + (lane & 15);
  int hi = lane >> 4;
  if (n < N) {
    float bv = bias ? bias[n] : 0.f;
#pragma unroll
    for (int r = 0; r < 8; ++r) {
      int m = row0 + r + 8 * hi;
      if (m < M) {
        float v = apply_act(acc[r] + bv, act);
        if (residual) v += residual[(size_t)m * N + n];
        C[(size_t)m * N + n] = v;
      }
    }
  }
}

// ---------------------------------------------------------------------------
// Fused MHA, f16 in/out: qkv (seq,256,384) [q|k|v each 128, head h at h*32].
// One wave per (qtile, head, seq). QK^T -> LDS softmax -> PV, all WMMA.
// ---------------------------------------------------------------------------
__launch_bounds__(32)
__global__ void attn_kernel(const _Float16* __restrict__ qkv, _Float16* __restrict__ out) {
  const int lane = threadIdx.x;
  const int l16 = lane & 15;
  const int hi = lane >> 4;
  const int hs = hi << 3;
  const int qt = blockIdx.x;   // 0..15
  const int h = blockIdx.y;    // 0..3
  const int seq = blockIdx.z;  // 0..255
  const _Float16* base = qkv + (size_t)seq * 256 * 384;
  __shared__ float sc[16][257];
  __shared__ _Float16 ph[16][256];
  const float scale = 0.17677669529663687f;  // 1/sqrt(32)

  v16h aq = fragA_h(base + (size_t)(qt * 16 + l16) * 384 + h * 32, 0, hs);
  for (int kt = 0; kt < 16; ++kt) {
    v16h bk = fragA_h(base + (size_t)(kt * 16 + l16) * 384 + 128 + h * 32, 0, hs);
    v8f s = {};
    s = WMMA_F16(aq, bk, s);
#pragma unroll
    for (int r = 0; r < 8; ++r) sc[r + 8 * hi][kt * 16 + l16] = s[r] * scale;
  }
  __syncthreads();
  if (lane < 16) {
    float mx = -3.4e38f;
    for (int j = 0; j < 256; ++j) mx = fmaxf(mx, sc[lane][j]);
    float sum = 0.f;
    for (int j = 0; j < 256; ++j) {
      float e = __expf(sc[lane][j] - mx);
      sc[lane][j] = e;
      sum += e;
    }
    float inv = 1.f / sum;
    for (int j = 0; j < 256; ++j) ph[lane][j] = (_Float16)(sc[lane][j] * inv);
  }
  __syncthreads();
#pragma unroll
  for (int nt = 0; nt < 2; ++nt) {
    v8f acc = {};
    for (int kt = 0; kt < 8; ++kt) {
      v16h ap = fragA_lds(&ph[l16][0], kt * 32, hs);
      v16h bv = fragB_rm_h(base + (size_t)(kt * 32) * 384 + 256 + h * 32 + nt * 16 + l16,
                           384, hs);
      acc = WMMA_F16(ap, bv, acc);
    }
#pragma unroll
    for (int r = 0; r < 8; ++r)
      out[(size_t)seq * 256 * 128 + (size_t)(qt * 16 + r + 8 * hi) * 128 + h * 32 +
          nt * 16 + l16] = (_Float16)acc[r];
  }
}

// ---------------------------------------------------------------------------
// LayerNorm: f32 in, f16 out; one wave per row (wave32 shuffle reductions).
// ---------------------------------------------------------------------------
__launch_bounds__(32)
__global__ void ln_kernel(const float* __restrict__ x, const float* __restrict__ g,
                          const float* __restrict__ b, _Float16* __restrict__ y, int D) {
  const int row = blockIdx.x;
  const int lane = threadIdx.x;
  const float* xr = x + (size_t)row * D;
  float s = 0.f;
  for (int i = lane; i < D; i += 32) s += xr[i];
#pragma unroll
  for (int off = 16; off; off >>= 1) s += __shfl_xor(s, off, 32);
  float mean = s / D;
  float v = 0.f;
  for (int i = lane; i < D; i += 32) {
    float d = xr[i] - mean;
    v += d * d;
  }
#pragma unroll
  for (int off = 16; off; off >>= 1) v += __shfl_xor(v, off, 32);
  float inv = rsqrtf(v / D + 1e-5f);
  _Float16* yr = y + (size_t)row * D;
  for (int i = lane; i < D; i += 32)
    yr[i] = (_Float16)((xr[i] - mean) * inv * g[i] + b[i]);
}

// ---------------------------------------------------------------------------
// Multi-scale conv1d over segments + BN + ReLU; reads tm_res (8,4224,4) with
// the segment gather fused in; writes (256,256,128) = (seq, L, C) in f32.
// ---------------------------------------------------------------------------
__global__ void msc_kernel(const float* __restrict__ tmres, const float* __restrict__ w3,
                           const float* __restrict__ b3, const float* __restrict__ w5,
                           const float* __restrict__ b5, const float* __restrict__ w7,
                           const float* __restrict__ b7, const float* __restrict__ bng,
                           const float* __restrict__ bnb, float* __restrict__ out) {
  int idx = blockIdx.x * blockDim.x + threadIdx.x;
  if (idx >= 256 * 256 * 128) return;
  int c = idx & 127;
  int l = (idx >> 7) & 255;
  int seq = idx >> 15;
  int bt = seq >> 5, s = seq & 31;
  const float* src = tmres + ((size_t)bt * 4224 + s * 128) * 4;  // (pos, ch)
  const float* w;
  float acc;
  int k;
  if (c < 42)      { k = 3; w = w3 + c * 12;         acc = b3[c]; }
  else if (c < 84) { k = 5; w = w5 + (c - 42) * 20;  acc = b5[c - 42]; }
  else             { k = 7; w = w7 + (c - 84) * 28;  acc = b7[c - 84]; }
  int half = k >> 1;
  for (int j = 0; j < k; ++j) {
    int p = l + j - half;
    if (p < 0 || p >= 256) continue;
    const float* xp = src + (size_t)p * 4;
    for (int ci = 0; ci < 4; ++ci) acc += w[ci * k + j] * xp[ci];
  }
  float vv = acc * 0.99999500003749968f * bng[c] + bnb[c];  // * 1/sqrt(1+1e-5)
  out[idx] = fmaxf(vv, 0.f);
}

// Mean over L: x (S, L, C) -> out (S, C)
__global__ void rowmean_kernel(const float* __restrict__ x, float* __restrict__ out,
                               int S_, int L_, int C_) {
  int idx = blockIdx.x * blockDim.x + threadIdx.x;
  if (idx >= S_ * C_) return;
  int c = idx % C_, s = idx / C_;
  const float* p = x + (size_t)s * L_ * C_ + c;
  float acc = 0.f;
  for (int l = 0; l < L_; ++l) acc += p[(size_t)l * C_];
  out[idx] = acc / (float)L_;
}

// spec conv1 3x3: spec (8,128,128,2) NHWC -> out (8,16,128,128) NCHW, ReLU.
__global__ void sp1_kernel(const float* __restrict__ spec, const float* __restrict__ w,
                           const float* __restrict__ b, float* __restrict__ out) {
  int idx = blockIdx.x * blockDim.x + threadIdx.x;
  if (idx >= 8 * 16 * 128 * 128) return;
  int x = idx & 127, y = (idx >> 7) & 127, co = (idx >> 14) & 15, bt = idx >> 18;
  float acc = b[co];
  for (int ci = 0; ci < 2; ++ci)
    for (int ky = 0; ky < 3; ++ky) {
      int yy = y + ky - 1;
      if (yy < 0 || yy >= 128) continue;
      for (int kx = 0; kx < 3; ++kx) {
        int xx = x + kx - 1;
        if (xx < 0 || xx >= 128) continue;
        acc += w[((co * 2 + ci) * 3 + ky) * 3 + kx] *
               spec[(((size_t)bt * 128 + yy) * 128 + xx) * 2 + ci];
      }
    }
  out[idx] = fmaxf(acc, 0.f);
}

// spec conv2 3x3 fused with spatial mean: h (8,16,128,128) -> out (8,128)
__launch_bounds__(256)
__global__ void sp2_mean_kernel(const float* __restrict__ h, const float* __restrict__ w,
                                const float* __restrict__ b, float* __restrict__ out) {
  int co = blockIdx.x & 127, bt = blockIdx.x >> 7;
  int t = threadIdx.x;
  float acc = 0.f;
  for (int pix = t; pix < 16384; pix += 256) {
    int x = pix & 127, y = pix >> 7;
    float v = 0.f;
    for (int ci = 0; ci < 16; ++ci) {
      const float* hp = h + ((size_t)bt * 16 + ci) * 16384;
      const float* wp = w + (size_t)(co * 16 + ci) * 9;
      for (int ky = 0; ky < 3; ++ky) {
        int yy = y + ky - 1;
        if (yy < 0 || yy >= 128) continue;
        for (int kx = 0; kx < 3; ++kx) {
          int xx = x + kx - 1;
          if (xx < 0 || xx >= 128) continue;
          v += wp[ky * 3 + kx] * hp[yy * 128 + xx];
        }
      }
    }
    acc += v;
  }
  __shared__ float red[256];
  red[t] = acc;
  __syncthreads();
  for (int st = 128; st; st >>= 1) {
    if (t < st) red[t] += red[t + st];
    __syncthreads();
  }
  if (t == 0) out[bt * 128 + co] = red[0] / 16384.f + b[co];
}

// fe (256,128) += sp[b] broadcast over the 32 segments of batch b.
__global__ void add_spec_kernel(float* __restrict__ fe, const float* __restrict__ sp) {
  int idx = blockIdx.x * blockDim.x + threadIdx.x;
  if (idx >= 256 * 128) return;
  int c = idx & 127;
  int bt = idx >> 12;
  fe[idx] += sp[bt * 128 + c];
}

// ---------------------------------------------------------------------------
// Gating head: summary = mean(f1 over Sn), w = softmax(relu([summary,snr]@g1)@g2)
// ---------------------------------------------------------------------------
__launch_bounds__(64)
__global__ void gate_kernel(const float* __restrict__ f1, const float* __restrict__ snr,
                            const float* __restrict__ g1w, const float* __restrict__ g1b,
                            const float* __restrict__ g2w, const float* __restrict__ g2b,
                            float* __restrict__ wout) {
  int bt = blockIdx.x, t = threadIdx.x;
  __shared__ float summ[128], hsh[64], lsh[3];
  for (int c = t; c < 128; c += 64) {
    float s = 0.f;
    for (int sg = 0; sg < 32; ++sg) s += f1[((size_t)bt * 32 + sg) * 128 + c];
    summ[c] = s * (1.f / 32.f);
  }
  __syncthreads();
  {
    float acc = g1b[t];
    for (int k = 0; k < 128; ++k) acc += summ[k] * g1w[k * 64 + t];
    acc += snr[bt] * g1w[128 * 64 + t];
    hsh[t] = fmaxf(acc, 0.f);
  }
  __syncthreads();
  if (t < 3) {
    float acc = g2b[t];
    for (int j = 0; j < 64; ++j) acc += hsh[j] * g2w[j * 3 + t];
    lsh[t] = acc;
  }
  __syncthreads();
  if (t == 0) {
    float mx = fmaxf(lsh[0], fmaxf(lsh[1], lsh[2]));
    float e0 = expf(lsh[0] - mx), e1 = expf(lsh[1] - mx), e2 = expf(lsh[2] - mx);
    float inv = 1.f / (e0 + e1 + e2);
    wout[bt * 3 + 0] = e0 * inv;
    wout[bt * 3 + 1] = e1 * inv;
    wout[bt * 3 + 2] = e2 * inv;
  }
}

// x0 = w0*f1 + w1*f2 + w2*f3 + snr proj + pos  -> (8,32,128)
__global__ void combine_kernel(const float* __restrict__ f1, const float* __restrict__ f2,
                               const float* __restrict__ f3, const float* __restrict__ wg,
                               const float* __restrict__ snr,
                               const float* __restrict__ snrw,
                               const float* __restrict__ snrb,
                               const float* __restrict__ pos, float* __restrict__ xg) {
  int idx = blockIdx.x * blockDim.x + threadIdx.x;
  if (idx >= 8 * 32 * 128) return;
  int c = idx & 127;
  int s = (idx >> 7) & 31;
  int bt = idx >> 12;
  float v = wg[bt * 3] * f1[idx] + wg[bt * 3 + 1] * f2[idx] + wg[bt * 3 + 2] * f3[idx];
  v += snr[bt] * snrw[c] + snrb[c] + pos[s * 128 + c];
  xg[idx] = v;
}

// ---------------------------------------------------------------------------
// GATv2 kernels. xl/xr are (B,Sn,4,64) = (B,Sn,256); logits (B,Sn,Sn,4).
// ---------------------------------------------------------------------------
__global__ void gat_logits_kernel(const float* __restrict__ xl,
                                  const float* __restrict__ xr,
                                  const float* __restrict__ att, float* __restrict__ lg) {
  int idx = blockIdx.x * blockDim.x + threadIdx.x;
  if (idx >= 8 * 32 * 32 * 4) return;
  int h = idx & 3;
  int s = (idx >> 2) & 31;
  int t = (idx >> 7) & 31;
  int bt = idx >> 12;
  const float* r = xr + (((size_t)bt * 32 + t) * 4 + h) * 64;
  const float* l = xl + (((size_t)bt * 32 + s) * 4 + h) * 64;
  const float* a = att + h * 64;
  float acc = 0.f;
  for (int d = 0; d < 64; ++d) {
    float e = r[d] + l[d];
    e = (e > 0.f) ? e : 0.2f * e;
    acc += e * a[d];
  }
  lg[idx] = acc;
}

__global__ void gat_softmax_kernel(float* __restrict__ lg) {
  int idx = blockIdx.x * blockDim.x + threadIdx.x;
  if (idx >= 8 * 32 * 4) return;
  int h = idx & 3;
  int t = (idx >> 2) & 31;
  int bt = idx >> 7;
  float* p = lg + ((size_t)bt * 32 + t) * 128 + h;  // stride 4 over s
  float mx = -3.4e38f;
  for (int s = 0; s < 32; ++s) mx = fmaxf(mx, p[s * 4]);
  float sum = 0.f;
  for (int s = 0; s < 32; ++s) {
    float e = __expf(p[s * 4] - mx);
    p[s * 4] = e;
    sum += e;
  }
  float inv = 1.f / sum;
  for (int s = 0; s < 32; ++s) p[s * 4] *= inv;
}

__global__ void gat_out_concat_kernel(const float* __restrict__ lg,
                                      const float* __restrict__ xl,
                                      const float* __restrict__ bias,
                                      float* __restrict__ go) {
  int idx = blockIdx.x * blockDim.x + threadIdx.x;
  if (idx >= 8 * 32 * 4 * 64) return;
  int d = idx & 63;
  int h = (idx >> 6) & 3;
  int t = (idx >> 8) & 31;
  int bt = idx >> 13;
  const float* al = lg + ((size_t)bt * 32 + t) * 128 + h;
  float acc = 0.f;
  for (int s = 0; s < 32; ++s)
    acc += al[s * 4] * xl[(((size_t)bt * 32 + s) * 4 + h) * 64 + d];
  go[(((size_t)bt * 32 + t) * 4 + h) * 64 + d] = acc + bias[h * 64 + d];
}

__global__ void gat_out_mean_kernel(const float* __restrict__ lg,
                                    const float* __restrict__ xl,
                                    const float* __restrict__ bias,
                                    float* __restrict__ go) {
  int idx = blockIdx.x * blockDim.x + threadIdx.x;
  if (idx >= 8 * 32 * 64) return;
  int d = idx & 63;
  int t = (idx >> 6) & 31;
  int bt = idx >> 11;
  float acc = 0.f;
  for (int h = 0; h < 4; ++h) {
    const float* al = lg + ((size_t)bt * 32 + t) * 128 + h;
    for (int s = 0; s < 32; ++s)
      acc += al[s * 4] * xl[(((size_t)bt * 32 + s) * 4 + h) * 64 + d];
  }
  go[((size_t)bt * 32 + t) * 64 + d] = acc * 0.25f + bias[d];
}

// GraphNorm + ReLU in-place over (8, 32, C); one thread per (b, c).
__global__ void graphnorm_kernel(float* __restrict__ x, const float* __restrict__ g,
                                 const float* __restrict__ b, const float* __restrict__ ms,
                                 int C) {
  int idx = blockIdx.x * blockDim.x + threadIdx.x;
  if (idx >= 8 * C) return;
  int c = idx % C, bt = idx / C;
  float* p = x + (size_t)bt * 32 * C + c;
  float mu = 0.f;
  for (int s = 0; s < 32; ++s) mu += p[(size_t)s * C];
  mu *= (1.f / 32.f);
  float m = ms[c] * mu;
  float var = 0.f;
  for (int s = 0; s < 32; ++s) {
    float d = p[(size_t)s * C] - m;
    var += d * d;
  }
  var *= (1.f / 32.f);
  float inv = rsqrtf(var + 1e-5f);
  for (int s = 0; s < 32; ++s) {
    float v = g[c] * (p[(size_t)s * C] - m) * inv + b[c];
    p[(size_t)s * C] = fmaxf(v, 0.f);
  }
}

// tanh scoring + top-16 selection + weighted sum; one block per batch.
__launch_bounds__(64)
__global__ void pool_topk_kernel(const float* __restrict__ x, const float* __restrict__ pw,
                                 const float* __restrict__ pb, float* __restrict__ g) {
  int bt = blockIdx.x, t = threadIdx.x;
  __shared__ float score[32];
  __shared__ float vals[16];
  __shared__ int tops[16];
  if (t < 32) {
    const float* p = x + ((size_t)bt * 32 + t) * 64;
    float s = 0.f;
    for (int c = 0; c < 64; ++c) s += p[c] * pw[c];
    score[t] = tanhf(s + pb[0]);
  }
  __syncthreads();
  if (t == 0) {
    float loc[32];
    for (int i = 0; i < 32; ++i) loc[i] = score[i];
    for (int k = 0; k < 16; ++k) {
      int bi = 0;
      float bv = loc[0];
      for (int i = 1; i < 32; ++i)
        if (loc[i] > bv) { bv = loc[i]; bi = i; }
      vals[k] = bv;
      tops[k] = bi;
      loc[bi] = -3.4e38f;
    }
  }
  __syncthreads();
  float acc = 0.f;
  for (int k = 0; k < 16; ++k) acc += vals[k] * x[((size_t)bt * 32 + tops[k]) * 64 + t];
  g[bt * 64 + t] = acc;
}

// ---------------------------------------------------------------------------
// Autoencoder conv1d (k=3, pad=1) over T=4224; first layer reads tm (NLC).
// ---------------------------------------------------------------------------
__global__ void ae_conv_kernel(const float* __restrict__ in, const float* __restrict__ w,
                               const float* __restrict__ bias, float* __restrict__ out,
                               int Cin, int Cout, int relu, int nlc) {
  const int T = 4224;
  int idx = blockIdx.x * blockDim.x + threadIdx.x;
  if (idx >= 8 * Cout * T) return;
  int t = idx % T;
  int co = (idx / T) % Cout;
  int bt = idx / (T * Cout);
  float acc = bias[co];
  const float* wp = w + (size_t)co * Cin * 3;
  for (int j = 0; j < 3; ++j) {
    int p = t + j - 1;
    if (p < 0 || p >= T) continue;
    if (nlc) {
      const float* xp = in + ((size_t)bt * T + p) * Cin;
      for (int ci = 0; ci < Cin; ++ci) acc += wp[ci * 3 + j] * xp[ci];
    } else {
      for (int ci = 0; ci < Cin; ++ci)
        acc += wp[ci * 3 + j] * in[((size_t)bt * Cin + ci) * T + p];
    }
  }
  out[idx] = relu ? fmaxf(acc, 0.f) : acc;
}

// tm_res = tm - recon (transposed); also writes recon_out into d_out.
__global__ void ae_post_kernel(const float* __restrict__ tm, const float* __restrict__ recon,
                               float* __restrict__ tmres, float* __restrict__ recon_out) {
  int idx = blockIdx.x * blockDim.x + threadIdx.x;
  if (idx >= 8 * 4224 * 4) return;
  int c = idx & 3;
  int rest = idx >> 2;
  int t = rest % 4224;
  int bt = rest / 4224;
  float rv = recon[((size_t)bt * 4 + c) * 4224 + t];
  recon_out[idx] = rv;
  tmres[idx] = tm[idx] - rv;
}

// ---------------------------------------------------------------------------
// Host side
// ---------------------------------------------------------------------------
struct Enc {
  const float *bn_b, *bn_g, *fc1_b, *fc1_w, *fc2_b, *fc2_w, *in_b, *in_w;
  const float *ln1_b, *ln1_g, *ln2_b, *ln2_g;
  const float *msc0_b, *msc0_w, *msc1_b, *msc1_w, *msc2_b, *msc2_w;
  const float *out_b, *out_w, *scat_b, *scat_w, *sp1_b, *sp1_w, *sp2_b, *sp2_w;
};

static Enc load_enc(void* const* d_in, int base) {
  Enc e;
  const float** f = (const float**)&e;
  for (int i = 0; i < 26; ++i) f[i] = (const float*)d_in[base + i];
  return e;
}

static inline void pack_b(hipStream_t st, const float* B, _Float16* out, int K, int N) {
  int tot = K * N;
  pack_b_kernel<<<(tot + 255) / 256, 256, 0, st>>>(B, out, K, N);
}

template <typename AT, typename OT>
static inline void gemm32p(hipStream_t st, const AT* A, const _Float16* Bp,
                           const float* bias, const float* res, OT* C, int M, int N,
                           int K, int act) {
  dim3 g(N / 32, M / 32);
  gemm32p_kernel<AT, OT><<<g, dim3(32), 0, st>>>(A, Bp, bias, res, C, M, N, K, act);
}

static inline void gemm_edge(hipStream_t st, const float* A, const float* B,
                             const float* bias, const float* res, float* C, int M, int N,
                             int K, int act) {
  dim3 g((N + 15) / 16, (M + 15) / 16);
  gemm_edge_kernel<<<g, dim3(32), 0, st>>>(A, B, bias, res, C, M, N, K, act);
}

struct EncWs {
  _Float16 *pwin, *pwout, *pwf1, *pwf2;   // packed weights
  _Float16 *qkv16, *bs16, *bh16;          // f16 activations
  float *bx, *scr32;                       // f32 activations / scratch
};

static void run_encoder(hipStream_t st, const Enc& e, const EncWs& w, const float* tmres,
                        const float* spec, const float* scat, float* fe, float* sp) {
  // Pack this encoder's weights to fragment-major f16.
  pack_b(st, e.in_w, w.pwin, 128, 384);
  pack_b(st, e.out_w, w.pwout, 128, 128);
  pack_b(st, e.fc1_w, w.pwf1, 128, 256);
  pack_b(st, e.fc2_w, w.pwf2, 256, 128);

  int tot = 256 * 256 * 128;
  msc_kernel<<<(tot + 255) / 256, 256, 0, st>>>(tmres, e.msc0_w, e.msc0_b, e.msc1_w,
                                                e.msc1_b, e.msc2_w, e.msc2_b, e.bn_g,
                                                e.bn_b, w.bx);
  ln_kernel<<<65536, 32, 0, st>>>(w.bx, e.ln1_g, e.ln1_b, w.bs16, 128);
  gemm32p<_Float16, _Float16>(st, w.bs16, w.pwin, e.in_b, nullptr, w.qkv16, 65536, 384,
                              128, ACT_NONE);
  attn_kernel<<<dim3(16, 4, 256), 32, 0, st>>>(w.qkv16, w.bs16);
  gemm32p<_Float16, float>(st, w.bs16, w.pwout, e.out_b, w.bx, w.bx, 65536, 128, 128,
                           ACT_NONE);
  ln_kernel<<<65536, 32, 0, st>>>(w.bx, e.ln2_g, e.ln2_b, w.bs16, 128);
  gemm32p<_Float16, _Float16>(st, w.bs16, w.pwf1, e.fc1_b, nullptr, w.bh16, 65536, 256,
                              128, ACT_GELU);
  gemm32p<_Float16, float>(st, w.bh16, w.pwf2, e.fc2_b, w.bx, w.bx, 65536, 128, 256,
                           ACT_NONE);
  rowmean_kernel<<<(256 * 128 + 255) / 256, 256, 0, st>>>(w.bx, fe, 256, 256, 128);
  gemm_edge(st, scat, e.scat_w, e.scat_b, fe, fe, 256, 128, 44, ACT_RELU);
  sp1_kernel<<<(8 * 16 * 128 * 128 + 255) / 256, 256, 0, st>>>(spec, e.sp1_w, e.sp1_b,
                                                               w.scr32);
  sp2_mean_kernel<<<1024, 256, 0, st>>>(w.scr32, e.sp2_w, e.sp2_b, sp);
  add_spec_kernel<<<(256 * 128 + 255) / 256, 256, 0, st>>>(fe, sp);
}

extern "C" void kernel_launch(void* const* d_in, const int* in_sizes, int n_in,
                              void* d_out, int out_size, void* d_ws, size_t ws_size,
                              hipStream_t stream) {
  (void)in_sizes; (void)n_in; (void)out_size; (void)ws_size;
#define FP(i) ((const float*)d_in[i])
  const float* tm = FP(0);
  const float* spec = FP(1);
  const float* scat = FP(2);
  const float* snr = FP(3);
  const float *d1b = FP(4), *d1w = FP(5), *d2b = FP(6), *d2w = FP(7);
  const float *e1b = FP(8), *e1w = FP(9), *e2b = FP(10), *e2w = FP(11);
  Enc fsk = load_enc(d_in, 12);
  const float *g1b = FP(38), *g1w = FP(39), *g2b = FP(40), *g2w = FP(41);
  const float *h1b = FP(60), *h1w = FP(61), *h2b = FP(62), *h2w = FP(63);
  const float *pool_b = FP(73), *pool_w = FP(74), *pos = FP(75);
  Enc psk = load_enc(d_in, 76);
  Enc qam = load_enc(d_in, 102);
  const float *snrb = FP(128), *snrw = FP(129);

  // Workspace carve (float units; all sizes multiples of 8 -> 32B alignment).
  // Total ~36.6M floats (~146 MB) -> fits in MI455X's 192MB L2.
  float* W = (float*)d_ws;
  size_t o = 0;
  EncWs ws;
  ws.qkv16 = (_Float16*)(W + o); o += (size_t)65536 * 384 / 2;  // 12,582,912
  ws.bx = W + o;                 o += (size_t)65536 * 128;      //  8,388,608
  ws.bs16 = (_Float16*)(W + o);  o += (size_t)65536 * 128 / 2;  //  4,194,304
  ws.bh16 = (_Float16*)(W + o);  o += (size_t)65536 * 256 / 2;  //  8,388,608
  ws.scr32 = W + o;              o += 2400000;                  // AE + spec scratch
  ws.pwin = (_Float16*)(W + o);  o += 128 * 384 / 2;
  ws.pwout = (_Float16*)(W + o); o += 128 * 128 / 2;
  ws.pwf1 = (_Float16*)(W + o);  o += 128 * 256 / 2;
  ws.pwf2 = (_Float16*)(W + o);  o += 256 * 128 / 2;
  _Float16* pwl = (_Float16*)(W + o); o += 256 * 256 / 2;
  _Float16* pwr = (_Float16*)(W + o); o += 256 * 256 / 2;
  float* tmres = W + o; o += 8 * 4224 * 4;
  float* f1b = W + o;   o += 256 * 128;
  float* f2b = W + o;   o += 256 * 128;
  float* f3b = W + o;   o += 256 * 128;
  float* spb = W + o;   o += 8 * 128;
  float* wgb = W + o;   o += 32;
  float* xg = W + o;    o += 8 * 32 * 256;
  float* xl = W + o;    o += 8 * 32 * 256;
  float* xr = W + o;    o += 8 * 32 * 256;
  float* lg = W + o;    o += 8 * 32 * 32 * 4;
  float* go = W + o;    o += 8 * 32 * 256;
  float* gb = W + o;    o += 8 * 64;
  float* hb = W + o;    o += 8 * 128;

  float* logits = (float*)d_out;            // (8,24)
  float* recon_out = (float*)d_out + 192;   // (8,4224,4)

  // ---- Phase A: denoising autoencoder (uses scr32 as scratch) ----
  float* ha = ws.scr32;
  float* hb2 = ws.scr32 + 1081344;
  float* recon = ws.scr32 + 2162688;
  int aetot = 8 * 32 * 4224;
  ae_conv_kernel<<<(aetot + 255) / 256, 256, 0, stream>>>(tm, e1w, e1b, ha, 4, 32, 1, 1);
  ae_conv_kernel<<<(aetot + 255) / 256, 256, 0, stream>>>(ha, e2w, e2b, hb2, 32, 32, 1, 0);
  ae_conv_kernel<<<(aetot + 255) / 256, 256, 0, stream>>>(hb2, d1w, d1b, ha, 32, 32, 1, 0);
  int aetot2 = 8 * 4 * 4224;
  ae_conv_kernel<<<(aetot2 + 255) / 256, 256, 0, stream>>>(ha, d2w, d2b, recon, 32, 4, 0, 0);
  ae_post_kernel<<<(8 * 4224 * 4 + 255) / 256, 256, 0, stream>>>(tm, recon, tmres,
                                                                 recon_out);

  // ---- Phase B: three transformer encoders (WMMA-heavy, f16 data path) ----
  run_encoder(stream, psk, ws, tmres, spec, scat, f1b, spb);
  run_encoder(stream, qam, ws, tmres, spec, scat, f2b, spb);
  run_encoder(stream, fsk, ws, tmres, spec, scat, f3b, spb);

  // ---- Phase C: gating + mixture ----
  gate_kernel<<<8, 64, 0, stream>>>(f1b, snr, g1w, g1b, g2w, g2b, wgb);
  combine_kernel<<<(8 * 32 * 128 + 255) / 256, 256, 0, stream>>>(f1b, f2b, f3b, wgb, snr,
                                                                 snrw, snrb, pos, xg);

  // ---- Phase D: 3x GATv2 + GraphNorm ----
  float* cur = xg;
  float* nxt = go;
  int inC = 128;
  for (int i = 0; i < 3; ++i) {
    const float* att = FP(42 + i * 6);
    const float* gbias = FP(43 + i * 6);
    const float* bl = FP(44 + i * 6);
    const float* br = FP(45 + i * 6);
    const float* wl = FP(46 + i * 6);
    const float* wr = FP(47 + i * 6);
    const float* nb = FP(64 + i * 3);
    const float* ng = FP(65 + i * 3);
    const float* nms = FP(66 + i * 3);
    pack_b(stream, wl, pwl, inC, 256);
    pack_b(stream, wr, pwr, inC, 256);
    gemm32p<float, float>(stream, cur, pwl, bl, nullptr, xl, 256, 256, inC, ACT_NONE);
    gemm32p<float, float>(stream, cur, pwr, br, nullptr, xr, 256, 256, inC, ACT_NONE);
    gat_logits_kernel<<<(8 * 32 * 32 * 4 + 255) / 256, 256, 0, stream>>>(xl, xr, att, lg);
    gat_softmax_kernel<<<(8 * 32 * 4 + 255) / 256, 256, 0, stream>>>(lg);
    int outC;
    if (i < 2) {
      gat_out_concat_kernel<<<(8 * 32 * 256 + 255) / 256, 256, 0, stream>>>(lg, xl, gbias,
                                                                            nxt);
      outC = 256;
    } else {
      gat_out_mean_kernel<<<(8 * 32 * 64 + 255) / 256, 256, 0, stream>>>(lg, xl, gbias,
                                                                         nxt);
      outC = 64;
    }
    graphnorm_kernel<<<(8 * outC + 255) / 256, 256, 0, stream>>>(nxt, ng, nb, nms, outC);
    float* t = cur; cur = nxt; nxt = t;
    inC = outC;
  }

  // ---- Phase E: attention pooling + classifier head ----
  pool_topk_kernel<<<8, 64, 0, stream>>>(cur, pool_w, pool_b, gb);
  gemm_edge(stream, gb, h1w, h1b, nullptr, hb, 8, 128, 64, ACT_RELU);
  gemm_edge(stream, hb, h2w, h2b, nullptr, logits, 8, 24, 128, ACT_NONE);
#undef FP
}
